// Block_60662118088963
// MI455X (gfx1250) — matrix operand order repaired and checked by reading, not compiled, not run on hardware
//
#include <hip/hip_runtime.h>

// Problem constants (match reference)
#define Bn 2
#define Tn 2048
#define Cn 1024
#define Hn 16
#define Dn 64
#define Mn (Bn * Tn)   // 4096 token rows

typedef __bf16 bf16_t;
typedef __attribute__((ext_vector_type(8)))  __bf16 v8bf;
typedef __attribute__((ext_vector_type(16))) __bf16 v16bf;
typedef __attribute__((ext_vector_type(8)))  float  v8f;

// ---------------------------------------------------------------------------
// WMMA helper: D = A(16x32 bf16) * B(32x16 bf16) + C(16x16 f32)
// ---------------------------------------------------------------------------
__device__ __forceinline__ v8f wmma_bf16(v16bf a, v16bf b, v8f c) {
  return __builtin_amdgcn_wmma_f32_16x16x32_bf16(
      /*neg_a=*/false, a, /*neg_b=*/false, b,
      /*c_mod=*/(short)0, c, /*reuse_a=*/false, /*reuse_b=*/false);
}

__device__ __forceinline__ v16bf cat8(v8bf lo, v8bf hi) {
  return __builtin_shufflevector(lo, hi, 0, 1, 2, 3, 4, 5, 6, 7,
                                 8, 9, 10, 11, 12, 13, 14, 15);
}
// A-fragment pattern: elements p[0..15]               (two aligned 16B loads)
__device__ __forceinline__ v16bf frag_gap(const bf16_t* p) {
  return cat8(*(const v8bf*)p, *(const v8bf*)(p + 8));
}
// B-fragment pattern: elements p[0..15]               (two aligned 16B loads)
__device__ __forceinline__ v16bf frag_cont(const bf16_t* p) {
  return cat8(*(const v8bf*)p, *(const v8bf*)(p + 8));
}

// Async copy: 16 bytes global -> LDS per active lane (ASYNCcnt-tracked).
__device__ __forceinline__ void async_b128(unsigned lds_byte_off,
                                           const void* gaddr) {
  asm volatile("global_load_async_to_lds_b128 %0, %1, off"
               :: "v"(lds_byte_off), "v"(gaddr) : "memory");
}
__device__ __forceinline__ void wait_async0() {
  asm volatile("s_wait_asynccnt 0" ::: "memory");
}

// Branchless erf (Abramowitz-Stegun 7.1.26, |err| < 1.5e-7) + exact GELU
__device__ __forceinline__ float erf_fast(float x) {
  float ax = fabsf(x);
  float t  = __frcp_rn(__fmaf_rn(0.3275911f, ax, 1.0f));
  float p  = t * __fmaf_rn(t, __fmaf_rn(t, __fmaf_rn(t,
                 __fmaf_rn(t, 1.061405429f, -1.453152027f),
                 1.421413741f), -0.284496736f), 0.254829592f);
  float r  = __fmaf_rn(-p, __expf(-ax * ax), 1.0f);
  return copysignf(r, x);
}
__device__ __forceinline__ float gelu_exact(float x) {
  return 0.5f * x * (1.0f + erf_fast(x * 0.70710678118654752f));
}

// ---------------------------------------------------------------------------
// fp32 [K][N] -> bf16 transposed [N][K]   (LDS-tiled 32x32)
// ---------------------------------------------------------------------------
__global__ void transpose_f32_bf16_kernel(const float* __restrict__ in,
                                          bf16_t* __restrict__ out,
                                          int K, int N) {
  __shared__ bf16_t tile[32][33];
  int n0 = blockIdx.x * 32, k0 = blockIdx.y * 32;
  int tx = threadIdx.x & 31, ty = threadIdx.x >> 5;   // 256 threads: ty 0..7
#pragma unroll
  for (int i = ty; i < 32; i += 8)
    tile[i][tx] = (bf16_t)in[(size_t)(k0 + i) * N + n0 + tx];
  __syncthreads();
#pragma unroll
  for (int i = ty; i < 32; i += 8)
    out[(size_t)(n0 + i) * K + k0 + tx] = tile[tx][i];
}

// ---------------------------------------------------------------------------
// Extract transposed V from qkv: vT[(b*H+h)*D + d][t]  (LDS-tiled 32x32)
// grid: (Tn/32, Dn/32, Bn*Hn)
// ---------------------------------------------------------------------------
__global__ void vtrans_kernel(const bf16_t* __restrict__ qkv,
                              bf16_t* __restrict__ vT) {
  __shared__ bf16_t tile[32][33];
  int t0 = blockIdx.x * 32, d0 = blockIdx.y * 32;
  int bh = blockIdx.z;
  int b = bh / Hn, h = bh % Hn;
  int tx = threadIdx.x & 31, ty = threadIdx.x >> 5;
  const bf16_t* src = qkv + (size_t)(b * Tn) * (3 * Cn) + 2 * Cn + h * Dn;
#pragma unroll
  for (int i = ty; i < 32; i += 8)
    tile[i][tx] = src[(size_t)(t0 + i) * (3 * Cn) + d0 + tx];
  __syncthreads();
  bf16_t* dst = vT + (size_t)bh * Dn * Tn;
#pragma unroll
  for (int i = ty; i < 32; i += 8)
    dst[(size_t)(d0 + i) * Tn + t0 + tx] = tile[tx][i];
}

// ---------------------------------------------------------------------------
// LayerNorm: f32 input rows -> bf16 output rows  (one block per row)
// ---------------------------------------------------------------------------
__device__ __forceinline__ float block_reduce_sum(float v) {
  __shared__ float sm[8];
#pragma unroll
  for (int o = 16; o > 0; o >>= 1) v += __shfl_xor(v, o, 32);
  int lane = threadIdx.x & 31, w = threadIdx.x >> 5;
  __syncthreads();
  if (lane == 0) sm[w] = v;
  __syncthreads();
  float r = 0.f;
#pragma unroll
  for (int i = 0; i < 8; ++i) r += sm[i];
  return r;
}

__global__ void layernorm_bf16_kernel(const float* __restrict__ x,
                                      const float* __restrict__ w,
                                      const float* __restrict__ bsc,
                                      bf16_t* __restrict__ out, int cols) {
  int row = blockIdx.x;
  const float* xr = x + (size_t)row * cols;
  float s = 0.f;
  for (int i = threadIdx.x; i < cols; i += blockDim.x) s += xr[i];
  float mu = block_reduce_sum(s) / cols;
  float v = 0.f;
  for (int i = threadIdx.x; i < cols; i += blockDim.x) {
    float d = xr[i] - mu;
    v += d * d;
  }
  float var = block_reduce_sum(v) / cols;
  float inv = rsqrtf(var + 1e-5f);
  bf16_t* orow = out + (size_t)row * cols;
  for (int i = threadIdx.x; i < cols; i += blockDim.x)
    orow[i] = (bf16_t)((xr[i] - mu) * inv * w[i] + bsc[i]);
}

// ---------------------------------------------------------------------------
// WMMA GEMM with async-to-LDS tile staging.
//   out = epilogue(A[M,K]bf16 @ W + bias) (+residual)
// W is given TRANSPOSED: WT[N][K] bf16.
// Block = 8 waves (2x4) covering 64(M) x 256(N); each wave computes 32x64.
// Per K-step (32): stage A(64x32) + B(256x32) tiles into LDS via
// GLOBAL_LOAD_ASYNC_TO_LDS_B128 (5 per thread), read fragments with
// ds_load_b128, then overlap next tile's async copy with 8 WMMAs.
// ---------------------------------------------------------------------------
#define GA_ELEMS (64 * 32)     // A tile elements in LDS
#define GB_ELEMS (256 * 32)    // B tile elements in LDS

template <int ACT, bool HAS_RES, bool OUT_F32>
__global__ void gemm_wmma_kernel(const bf16_t* __restrict__ A,
                                 const bf16_t* __restrict__ WT,
                                 const float* __restrict__ bias,
                                 const float* __restrict__ residual,
                                 float* __restrict__ outf,
                                 bf16_t* __restrict__ outb,
                                 int M, int N, int K) {
  // Single __shared__ object => allocated at LDS offset 0.
  __shared__ bf16_t sbuf[GA_ELEMS + GB_ELEMS];   // 20 KB

  int tid  = threadIdx.x;
  int lane = tid & 31;
  int wave = tid >> 5;
  int wm = wave & 1, wn = wave >> 1;             // 2 x 4 wave grid
  int rowBlk = blockIdx.y * 64;
  int colBlk = blockIdx.x * 256;

  // Staging roles:
  //   A tile: row = tid>>2 (64 rows), chunk = tid&3 (4 x 16B per 64B row)
  //   B tile: col = tid (256 WT rows), 4 x 16B chunks per 64B row
  int arow = tid >> 2, achk = tid & 3;
  const bf16_t* gA = A + (size_t)(rowBlk + arow) * K + achk * 8;
  const bf16_t* gB = WT + (size_t)(colBlk + tid) * K;
  unsigned lA = (unsigned)((arow * 32 + achk * 8) * sizeof(bf16_t));
  unsigned lB = (unsigned)((GA_ELEMS + tid * 32) * sizeof(bf16_t));

  auto issue_tile = [&](int k0) {
    async_b128(lA, gA + k0);
#pragma unroll
    for (int c = 0; c < 4; ++c)
      async_b128(lB + c * 16, gB + k0 + c * 8);
  };

  int r   = lane & 15;
  int kbA = (lane < 16) ? 0 : 16;                // A frag K base
  int kbB = (lane < 16) ? 0 : 16;                // B frag K base
  int n   = lane & 15;

  const bf16_t* Abuf = sbuf;
  const bf16_t* Bbuf = sbuf + GA_ELEMS;

  v8f acc[2][4] = {};
  issue_tile(0);
  for (int k0 = 0; k0 < K; k0 += 32) {
    wait_async0();                 // this wave's tile portion is in LDS
    __syncthreads();               // => every wave's portion is in LDS

    // Fragment reads from LDS (ds_load_b128 pairs)
    v16bf a0 = frag_gap(Abuf + (wm * 32 + r) * 32 + kbA);
    v16bf a1 = frag_gap(Abuf + (wm * 32 + 16 + r) * 32 + kbA);
    v16bf bq[4];
#pragma unroll
    for (int j = 0; j < 4; ++j)
      bq[j] = frag_cont(Bbuf + (wn * 64 + 16 * j + n) * 32 + kbB);

    asm volatile("s_wait_dscnt 0" ::: "memory");  // frags now in VGPRs
    __syncthreads();               // all waves done reading the tile

    if (k0 + 32 < K) issue_tile(k0 + 32);   // async copy overlaps WMMAs

#pragma unroll
    for (int j = 0; j < 4; ++j) {
      acc[0][j] = wmma_bf16(a0, bq[j], acc[0][j]);
      acc[1][j] = wmma_bf16(a1, bq[j], acc[1][j]);
    }
  }

  int row0 = rowBlk + wm * 32;
  int col0 = colBlk + wn * 64;
  int rb = (lane < 16) ? 0 : 8;   // C/D rows: lanes 0-15 hold M=0..7, 16-31 8..15
#pragma unroll
  for (int mi = 0; mi < 2; ++mi) {
#pragma unroll
    for (int j = 0; j < 4; ++j) {
      int c = col0 + 16 * j + n;
      float bv = bias[c];
#pragma unroll
      for (int v = 0; v < 8; ++v) {
        size_t idx = (size_t)(row0 + mi * 16 + rb + v) * N + c;
        float xv = acc[mi][j][v] + bv;
        if (ACT == 1) xv = gelu_exact(xv);
        if (HAS_RES) xv += residual[idx];
        if (OUT_F32) outf[idx] = xv;
        else         outb[idx] = (bf16_t)xv;
      }
    }
  }
}

// ---------------------------------------------------------------------------
// Flash-style causal attention.
// qkv: [Mn, 3*Cn] bf16 (Q,K rows, stride 3*Cn, head offset h*Dn).
// vT : [(b*H+h)*Dn + d][Tn] bf16 (transposed V for contiguous B-fragments).
// One wave per 16-row q tile; 4 waves per block; streams 32-wide k chunks.
// ---------------------------------------------------------------------------
__global__ void attention_kernel(const bf16_t* __restrict__ qkv,
                                 const bf16_t* __restrict__ vT,
                                 bf16_t* __restrict__ y) {
  __shared__ bf16_t Pl[4][16][32];          // per-wave P staging (C->A relayout)

  int lane = threadIdx.x & 31;
  int wave = threadIdx.x >> 5;
  int task = blockIdx.x * 4 + wave;         // Bn*Hn*(Tn/16) tasks
  int qt = task & (Tn / 16 - 1);
  int bh = task / (Tn / 16);
  int b = bh / Hn, h = bh % Hn;
  int q0 = qt * 16;
  const long ld = 3 * Cn;

  const bf16_t* Q  = qkv + (size_t)(b * Tn) * ld + h * Dn;
  const bf16_t* Km = Q + Cn;
  const bf16_t* Vt = vT + (size_t)bh * Dn * Tn;

  int n   = lane & 15;
  int rb  = (lane < 16) ? 0 : 8;
  int kbA = (lane < 16) ? 0 : 16;
  int kbB = (lane < 16) ? 0 : 16;

  // Q fragments (16 x 64 = two 16x32 A fragments), loaded once.
  v16bf qf[2];
  {
    const bf16_t* qp = Q + (size_t)(q0 + n) * ld + kbA;
    qf[0] = frag_gap(qp);
    qf[1] = frag_gap(qp + 32);
  }

  const float scale = 0.125f;               // 1/sqrt(64)
  const float NEG = -3.0e38f;

  float m_i[8], l_i[8];
  v8f o[4] = {};
#pragma unroll
  for (int v = 0; v < 8; ++v) { m_i[v] = NEG; l_i[v] = 0.f; }

  int jmax = (q0 + 15) / 32;                // last 32-wide chunk touching diagonal
  for (int j = 0; j <= jmax; ++j) {
    int k0 = j * 32;

    // ---- S = Q * K^T for k columns [k0, k0+32) : two 16x16 tiles ----
    v8f s0 = {}, s1 = {};
#pragma unroll
    for (int c = 0; c < 2; ++c) {           // K dim (Dn=64) in 2 chunks of 32
      // B[kk][n] = Km[(k col) * ld + d], contiguous along d
      v16bf b0 = frag_cont(Km + (size_t)(k0 + n) * ld + 32 * c + kbB);
      v16bf b1 = frag_cont(Km + (size_t)(k0 + 16 + n) * ld + 32 * c + kbB);
      s0 = wmma_bf16(qf[c], b0, s0);
      s1 = wmma_bf16(qf[c], b1, s1);
    }

    // ---- scale + causal mask + tile row-max ----
    float pm[8];
#pragma unroll
    for (int v = 0; v < 8; ++v) {
      int qg  = q0 + rb + v;
      float a0 = (k0 + n      <= qg) ? s0[v] * scale : NEG;
      float a1 = (k0 + 16 + n <= qg) ? s1[v] * scale : NEG;
      s0[v] = a0; s1[v] = a1;
      pm[v] = fmaxf(a0, a1);
    }
#pragma unroll
    for (int msk = 8; msk >= 1; msk >>= 1)   // stays inside each 16-lane half
#pragma unroll
      for (int v = 0; v < 8; ++v) pm[v] = fmaxf(pm[v], __shfl_xor(pm[v], msk, 32));

    // ---- online softmax update ----
    float alpha[8], rs[8];
#pragma unroll
    for (int v = 0; v < 8; ++v) {
      float mn = fmaxf(m_i[v], pm[v]);
      alpha[v] = __expf(m_i[v] - mn);
      m_i[v] = mn;
    }
#pragma unroll
    for (int v = 0; v < 8; ++v) {
      s0[v] = __expf(s0[v] - m_i[v]);
      s1[v] = __expf(s1[v] - m_i[v]);
      rs[v] = s0[v] + s1[v];
    }
#pragma unroll
    for (int msk = 8; msk >= 1; msk >>= 1)
#pragma unroll
      for (int v = 0; v < 8; ++v) rs[v] += __shfl_xor(rs[v], msk, 32);
#pragma unroll
    for (int v = 0; v < 8; ++v) l_i[v] = l_i[v] * alpha[v] + rs[v];
#pragma unroll
    for (int d = 0; d < 4; ++d)
#pragma unroll
      for (int v = 0; v < 8; ++v) o[d][v] *= alpha[v];

    // ---- re-layout P (C layout -> A layout) through LDS ----
#pragma unroll
    for (int v = 0; v < 8; ++v) {
      Pl[wave][rb + v][n]      = (bf16_t)s0[v];
      Pl[wave][rb + v][n + 16] = (bf16_t)s1[v];
    }
    asm volatile("s_wait_dscnt 0" ::: "memory");   // per-wave LDS RAW fence

    v16bf pa = frag_gap(&Pl[wave][n][0] + kbA);

    // ---- O += P(16x32) * V(32x64): B[kk][n] = Vt[(d0+n)*Tn + k0+kk] ----
#pragma unroll
    for (int d = 0; d < 4; ++d) {
      v16bf bv = frag_cont(Vt + (size_t)(d * 16 + n) * Tn + k0 + kbB);
      o[d] = wmma_bf16(pa, bv, o[d]);
    }
  }

  // ---- normalize and write y[Mn, Cn] bf16 ----
  float inv[8];
#pragma unroll
  for (int v = 0; v < 8; ++v) inv[v] = 1.0f / l_i[v];
#pragma unroll
  for (int d = 0; d < 4; ++d)
#pragma unroll
    for (int v = 0; v < 8; ++v) {
      size_t rr = (size_t)(b * Tn + q0 + rb + v);
      y[rr * Cn + h * Dn + d * 16 + n] = (bf16_t)(o[d][v] * inv[v]);
    }
}

// ---------------------------------------------------------------------------
// Host launcher
// ---------------------------------------------------------------------------
extern "C" void kernel_launch(void* const* d_in, const int* in_sizes, int n_in,
                              void* d_out, int out_size, void* d_ws, size_t ws_size,
                              hipStream_t stream) {
  const float* x      = (const float*)d_in[0];
  const float* ln1_w  = (const float*)d_in[1];
  const float* ln1_b  = (const float*)d_in[2];
  const float* W_attn = (const float*)d_in[3];
  const float* b_attn = (const float*)d_in[4];
  const float* W_proj = (const float*)d_in[5];
  const float* b_proj = (const float*)d_in[6];
  const float* ln2_w  = (const float*)d_in[7];
  const float* ln2_b  = (const float*)d_in[8];
  const float* W_fc   = (const float*)d_in[9];
  const float* b_fc   = (const float*)d_in[10];
  const float* W_fc2  = (const float*)d_in[11];
  const float* b_fc2  = (const float*)d_in[12];
  float* out = (float*)d_out;

  // Workspace layout (deterministic each call)
  char* ws = (char*)d_ws;
  size_t off = 0;
  auto take = [&](size_t bytes) -> char* {
    char* p = ws + off;
    off += (bytes + 255) & ~(size_t)255;
    return p;
  };
  bf16_t* wAT  = (bf16_t*)take((size_t)3 * Cn * Cn * sizeof(bf16_t));   // [3C][C]
  bf16_t* wPT  = (bf16_t*)take((size_t)Cn * Cn * sizeof(bf16_t));       // [C][C]
  bf16_t* wFT  = (bf16_t*)take((size_t)4 * Cn * Cn * sizeof(bf16_t));   // [4C][C]
  bf16_t* wF2T = (bf16_t*)take((size_t)Cn * 4 * Cn * sizeof(bf16_t));   // [C][4C]
  bf16_t* hb   = (bf16_t*)take((size_t)Mn * Cn * sizeof(bf16_t));       // ln1 out
  bf16_t* qkv  = (bf16_t*)take((size_t)Mn * 3 * Cn * sizeof(bf16_t));
  bf16_t* vT   = (bf16_t*)take((size_t)Bn * Hn * Dn * Tn * sizeof(bf16_t));
  bf16_t* yb   = (bf16_t*)take((size_t)Mn * Cn * sizeof(bf16_t));       // attn out
  float*  x1   = (float*)take((size_t)Mn * Cn * sizeof(float));         // x + proj
  bf16_t* h2   = (bf16_t*)take((size_t)Mn * Cn * sizeof(bf16_t));       // ln2 out
  bf16_t* afc  = (bf16_t*)take((size_t)Mn * 4 * Cn * sizeof(bf16_t));   // gelu(fc)

  // Weight transpose+convert: in f32 [K][N] -> out bf16 [N][K]
  auto tcvt = [&](const float* src, bf16_t* dst, int K, int N) {
    transpose_f32_bf16_kernel<<<dim3(N / 32, K / 32), 256, 0, stream>>>(src, dst, K, N);
  };
  tcvt(W_attn, wAT,  Cn, 3 * Cn);
  tcvt(W_proj, wPT,  Cn, Cn);
  tcvt(W_fc,   wFT,  Cn, 4 * Cn);
  tcvt(W_fc2,  wF2T, 4 * Cn, Cn);

  // ln1
  layernorm_bf16_kernel<<<Mn, 256, 0, stream>>>(x, ln1_w, ln1_b, hb, Cn);

  // qkv = ln1(x) @ W_attn + b_attn         [4096 x 3072], bf16 out
  gemm_wmma_kernel<0, false, false><<<dim3(3 * Cn / 256, Mn / 64), 256, 0, stream>>>(
      hb, wAT, b_attn, nullptr, nullptr, qkv, Mn, 3 * Cn, Cn);

  // transposed V extraction
  vtrans_kernel<<<dim3(Tn / 32, Dn / 32, Bn * Hn), 256, 0, stream>>>(qkv, vT);

  // causal flash attention -> yb [4096 x 1024]
  attention_kernel<<<(Bn * Hn * (Tn / 16)) / 4, 128, 0, stream>>>(qkv, vT, yb);

  // x1 = x + yb @ W_proj + b_proj          [4096 x 1024], f32 out + residual
  gemm_wmma_kernel<0, true, true><<<dim3(Cn / 256, Mn / 64), 256, 0, stream>>>(
      yb, wPT, b_proj, x, x1, nullptr, Mn, Cn, Cn);

  // ln2
  layernorm_bf16_kernel<<<Mn, 256, 0, stream>>>(x1, ln2_w, ln2_b, h2, Cn);

  // afc = gelu(ln2(x1) @ W_fc + b_fc)      [4096 x 4096], bf16 out
  gemm_wmma_kernel<1, false, false><<<dim3(4 * Cn / 256, Mn / 64), 256, 0, stream>>>(
      h2, wFT, b_fc, nullptr, nullptr, afc, Mn, 4 * Cn, Cn);

  // out = x1 + afc @ W_fc2 + b_fc2         [4096 x 1024], f32 out + residual
  gemm_wmma_kernel<0, true, true><<<dim3(Cn / 256, Mn / 64), 256, 0, stream>>>(
      afc, wF2T, b_fc2, x1, out, nullptr, Mn, Cn, 4 * Cn);
}
